// mLSTMCell_43301860278403
// MI455X (gfx1250) — matrix operand order repaired and checked by reading, not compiled, not run on hardware
//
#include <hip/hip_runtime.h>
#include <hip/hip_bf16.h>

// ---------------------------------------------------------------------------
// mLSTM cell for MI455X (gfx1250, wave32).
//   Phase 0: stream-convert X, h, and weights fp32 -> bf16 (one pass, ~2.4us).
//   Phase A: WMMA bf16 GEMMs (pure b128-load + v_wmma inner loop).
//   Phase B: streaming C update + fused q^T C readout (TH_NT, ~22us floor).
// ---------------------------------------------------------------------------

typedef __attribute__((ext_vector_type(16))) __bf16 v16bf;
typedef __attribute__((ext_vector_type(8)))  __bf16 v8bf;
typedef __attribute__((ext_vector_type(8)))  float  v8f;
typedef __attribute__((ext_vector_type(4)))  float  f4;

#define Bsz 64
#define Hdim 1024

// ---- Phase 0: fp32 -> bf16 streaming convert (8 elems / thread) ------------
__global__ void __launch_bounds__(256)
cvt_bf16_kernel(const float* __restrict__ src, __bf16* __restrict__ dst) {
    const size_t i = ((size_t)blockIdx.x * 256 + threadIdx.x) * 8;
    f4 a = *(const f4*)(src + i);
    f4 b = *(const f4*)(src + i + 4);
    v8bf o;
    o[0] = (__bf16)a.x; o[1] = (__bf16)a.y; o[2] = (__bf16)a.z; o[3] = (__bf16)a.w;
    o[4] = (__bf16)b.x; o[5] = (__bf16)b.y; o[6] = (__bf16)b.z; o[7] = (__bf16)b.w;
    *(v8bf*)(dst + i) = o;
}

// ---- bf16 fragment loaders (single-issue b128 loads, ISA 16-bit layouts) ---

// A 16x32 (MxK): lane m=lane&15; K groups {hs*8+0..7} and {16+hs*8+0..7}.
__device__ __forceinline__ v16bf load_a_bf(const __bf16* base, int ld,
                                           int row0, int kk, int lane) {
    const int m  = lane & 15;
    const int hs = lane >> 4;
    const __bf16* p = base + (size_t)(row0 + m) * ld + kk + hs * 8;
    union { v16bf v; v8bf h[2]; } u;
    u.h[0] = *(const v8bf*)p;
    u.h[1] = *(const v8bf*)(p + 16);
    return u.v;
}

// B 32x16 (KxN), B[k][n] = W[n][k]: lane n=lane&15 is the column,
// K groups {hs*8+0..7} and {16+hs*8+0..7} (same as A).
__device__ __forceinline__ v16bf load_b_bf(const __bf16* W, int ld,
                                           int ncol0, int kk, int lane) {
    const int n  = lane & 15;
    const int hs = lane >> 4;
    const __bf16* p = W + (size_t)(ncol0 + n) * ld + kk + hs * 8;
    union { v16bf v; v8bf h[2]; } u;
    u.h[0] = *(const v8bf*)p;
    u.h[1] = *(const v8bf*)(p + 16);
    return u.v;
}

// ---- Kernel A1: gates = X Wih^T + h Whh^T + bias; exp/exp/sigmoid ----------
// 48 blocks x 128 threads; each wave owns a 16-wide N tile, full M=64.
__global__ void __launch_bounds__(128)
gates_wmma_kernel(const __bf16* __restrict__ X, const __bf16* __restrict__ Hst,
                  const __bf16* __restrict__ Wih, const __bf16* __restrict__ Whh,
                  const float* __restrict__ bias,
                  float* __restrict__ iexp, float* __restrict__ fexp,
                  float* __restrict__ ogate) {
    const int lane  = threadIdx.x & 31;
    const int wv    = threadIdx.x >> 5;
    const int ntile = blockIdx.x * 4 + wv;      // 0..191
    const int ncol0 = ntile * 16;

    v8f acc[4] = {};
    #pragma unroll 2
    for (int kk = 0; kk < Hdim; kk += 32) {
        const v16bf bih = load_b_bf(Wih, Hdim, ncol0, kk, lane);
        const v16bf bhh = load_b_bf(Whh, Hdim, ncol0, kk, lane);
        #pragma unroll
        for (int mt = 0; mt < 4; ++mt) {
            const v16bf ax = load_a_bf(X,   Hdim, mt * 16, kk, lane);
            acc[mt] = __builtin_amdgcn_wmma_f32_16x16x32_bf16(
                false, ax, false, bih, (short)0, acc[mt], false, false);
            const v16bf ah = load_a_bf(Hst, Hdim, mt * 16, kk, lane);
            acc[mt] = __builtin_amdgcn_wmma_f32_16x16x32_bf16(
                false, ah, false, bhh, (short)0, acc[mt], false, false);
        }
    }

    const int hs   = lane >> 4;
    const int col  = ncol0 + (lane & 15);       // 0..3071
    const float bv = bias[col];
    const int g    = col >> 10;                 // uniform per wave
    const int hcol = col & (Hdim - 1);
    float* outp = (g == 0) ? iexp : (g == 1) ? fexp : ogate;
    #pragma unroll
    for (int mt = 0; mt < 4; ++mt) {
        #pragma unroll
        for (int i = 0; i < 8; ++i) {
            const int brow = mt * 16 + hs * 8 + i;
            const float val = acc[mt][i] + bv;
            const float r = (g == 2) ? 1.0f / (1.0f + __expf(-val)) : __expf(val);
            outp[brow * Hdim + hcol] = r;
        }
    }
}

// ---- Kernel A2: q/k/v = X W{q,k,v}^T + b -----------------------------------
__global__ void __launch_bounds__(128)
qkv_wmma_kernel(const __bf16* __restrict__ X,
                const __bf16* __restrict__ Wq, const float* __restrict__ bq,
                const __bf16* __restrict__ Wk, const float* __restrict__ bk,
                const __bf16* __restrict__ Wv, const float* __restrict__ bv_,
                float* __restrict__ qo, float* __restrict__ ko,
                float* __restrict__ vo) {
    const int lane  = threadIdx.x & 31;
    const int wv    = threadIdx.x >> 5;
    const int ntile = blockIdx.x * 4 + wv;      // 0..191
    const int which = ntile >> 6;               // 0=q,1=k,2=v (uniform per wave)
    const int ncol0 = (ntile & 63) * 16;

    const __bf16* W  = (which == 0) ? Wq : (which == 1) ? Wk : Wv;
    const float*  bb = (which == 0) ? bq : (which == 1) ? bk : bv_;
    float* outp      = (which == 0) ? qo : (which == 1) ? ko : vo;

    v8f acc[4] = {};
    #pragma unroll 2
    for (int kk = 0; kk < Hdim; kk += 32) {
        const v16bf bf = load_b_bf(W, Hdim, ncol0, kk, lane);
        #pragma unroll
        for (int mt = 0; mt < 4; ++mt) {
            const v16bf ax = load_a_bf(X, Hdim, mt * 16, kk, lane);
            acc[mt] = __builtin_amdgcn_wmma_f32_16x16x32_bf16(
                false, ax, false, bf, (short)0, acc[mt], false, false);
        }
    }

    const int hs   = lane >> 4;
    const int hcol = ncol0 + (lane & 15);
    const float bvv = bb[hcol];
    #pragma unroll
    for (int mt = 0; mt < 4; ++mt) {
        #pragma unroll
        for (int i = 0; i < 8; ++i) {
            const int brow = mt * 16 + hs * 8 + i;
            outp[brow * Hdim + hcol] = acc[mt][i] + bvv;
        }
    }
}

// ---- Kernel B: streaming C update + fused partial readout ------------------
// C_new[b,r,c] = f[b,r]*C[b,r,c] + (i[b,r]*v[b,r])*k[b,c]
// hpart[b,rc,c] = sum_{r in chunk} q[b,r]*C_new[b,r,c]
__global__ void __launch_bounds__(256)
c_update_kernel(const float* __restrict__ C,
                const float* __restrict__ iexp, const float* __restrict__ fexp,
                const float* __restrict__ q, const float* __restrict__ k,
                const float* __restrict__ v,
                float* __restrict__ Cnew, float* __restrict__ hpart) {
    __shared__ float sF[256], sIV[256], sQ[256];
    const int b   = blockIdx.y;
    const int r0  = blockIdx.x * 256;
    const int tid = threadIdx.x;

    const int rowidx = b * Hdim + r0 + tid;
    sF[tid]  = fexp[rowidx];
    sIV[tid] = iexp[rowidx] * v[rowidx];
    sQ[tid]  = q[rowidx];
    __syncthreads();

    const int col0 = tid * 4;
    const f4 kc = *(const f4*)(k + b * Hdim + col0);
    f4 hacc = (f4)0.0f;

    const size_t base = ((size_t)b * Hdim + r0) * Hdim + col0;
    const f4* __restrict__ cin  = (const f4*)(C + base);
    f4* __restrict__       cout = (f4*)(Cnew + base);

    #pragma unroll 4
    for (int rr = 0; rr < 256; ++rr) {
        const f4 c = __builtin_nontemporal_load(cin + (size_t)rr * (Hdim / 4));
        const float f  = sF[rr];
        const float iv = sIV[rr];
        const float qr = sQ[rr];
        f4 cn;
        cn.x = f * c.x + iv * kc.x;
        cn.y = f * c.y + iv * kc.y;
        cn.z = f * c.z + iv * kc.z;
        cn.w = f * c.w + iv * kc.w;
        __builtin_nontemporal_store(cn, cout + (size_t)rr * (Hdim / 4));
        hacc.x += qr * cn.x;
        hacc.y += qr * cn.y;
        hacc.z += qr * cn.z;
        hacc.w += qr * cn.w;
    }
    *(f4*)(hpart + ((size_t)(b * 4 + blockIdx.x)) * Hdim + col0) = hacc;
}

// ---- Kernel C: reduce partials, apply output gate --------------------------
__global__ void __launch_bounds__(256)
h_reduce_kernel(const float* __restrict__ hpart, const float* __restrict__ ogate,
                float* __restrict__ hout) {
    const int idx = blockIdx.x * 256 + threadIdx.x;   // 0..65535
    const int b = idx >> 10;
    const int c = idx & (Hdim - 1);
    const size_t p = (size_t)b * 4 * Hdim + c;
    float s = hpart[p] + hpart[p + Hdim] + hpart[p + 2 * Hdim] + hpart[p + 3 * Hdim];
    hout[idx] = ogate[idx] * s;
}

// ---------------------------------------------------------------------------
extern "C" void kernel_launch(void* const* d_in, const int* in_sizes, int n_in,
                              void* d_out, int out_size, void* d_ws, size_t ws_size,
                              hipStream_t stream) {
    const float* X    = (const float*)d_in[0];   // input  [64,1024]
    const float* Hst  = (const float*)d_in[1];   // h      [64,1024]
    const float* C    = (const float*)d_in[2];   // C      [64,1024,1024]
    const float* Wih  = (const float*)d_in[3];   // [3072,1024]
    const float* Whh  = (const float*)d_in[4];   // [3072,1024]
    const float* bias = (const float*)d_in[5];   // [3072]
    const float* Wq   = (const float*)d_in[6];
    const float* bq   = (const float*)d_in[7];
    const float* Wk   = (const float*)d_in[8];
    const float* bk   = (const float*)d_in[9];
    const float* Wv   = (const float*)d_in[10];
    const float* bv   = (const float*)d_in[11];

    // float scratch
    float* ws   = (float*)d_ws;
    float* iexp = ws;                       // 64*1024 each
    float* fexp = ws + 1 * Bsz * Hdim;
    float* og   = ws + 2 * Bsz * Hdim;
    float* qb   = ws + 3 * Bsz * Hdim;
    float* kb   = ws + 4 * Bsz * Hdim;
    float* vb   = ws + 5 * Bsz * Hdim;
    float* hprt = ws + 6 * Bsz * Hdim;      // 64*4*1024
    // bf16 scratch (starts 16B-aligned after 10*64K floats)
    __bf16* bfbase = (__bf16*)(ws + 10 * Bsz * Hdim);
    __bf16* Xb   = bfbase;                           // 65536
    __bf16* Hb   = Xb   + Bsz * Hdim;                // 65536
    __bf16* Wihb = Hb   + Bsz * Hdim;                // 3*1024*1024
    __bf16* Whhb = Wihb + 3 * Hdim * Hdim;           // 3*1024*1024
    __bf16* Wqb  = Whhb + 3 * Hdim * Hdim;           // 1024*1024
    __bf16* Wkb  = Wqb  + Hdim * Hdim;
    __bf16* Wvb  = Wkb  + Hdim * Hdim;

    float* hout = (float*)d_out;            // [64,1024]
    float* Cout = hout + Bsz * Hdim;        // [64,1024,1024]

    // Phase 0: fp32 -> bf16 (grid = n/8/256)
    cvt_bf16_kernel<<<  32, 256, 0, stream>>>(X,   Xb);
    cvt_bf16_kernel<<<  32, 256, 0, stream>>>(Hst, Hb);
    cvt_bf16_kernel<<<1536, 256, 0, stream>>>(Wih, Wihb);
    cvt_bf16_kernel<<<1536, 256, 0, stream>>>(Whh, Whhb);
    cvt_bf16_kernel<<< 512, 256, 0, stream>>>(Wq,  Wqb);
    cvt_bf16_kernel<<< 512, 256, 0, stream>>>(Wk,  Wkb);
    cvt_bf16_kernel<<< 512, 256, 0, stream>>>(Wv,  Wvb);

    // Phase A: WMMA GEMMs
    gates_wmma_kernel<<<48, 128, 0, stream>>>(Xb, Hb, Wihb, Whhb, bias,
                                              iexp, fexp, og);
    qkv_wmma_kernel<<<48, 128, 0, stream>>>(Xb, Wqb, bq, Wkb, bk, Wvb, bv,
                                            qb, kb, vb);

    // Phase B: streaming update + fused readout
    c_update_kernel<<<dim3(4, Bsz), 256, 0, stream>>>(C, iexp, fexp, qb, kb, vb,
                                                      Cout, hprt);
    h_reduce_kernel<<<256, 256, 0, stream>>>(hprt, og, hout);
}
